// ConvTreeGRUCell_59141699666271
// MI455X (gfx1250) — compile-verified, matrix operand order checked by reading, not run on hardware
//
#include <hip/hip_runtime.h>
#include <hip/hip_bf16.h>

// ---------------------------------------------------------------------------
// ConvTreeGRU cell for MI455X (gfx1250, wave32, WMMA bf16).
// Implicit-GEMM 3x3 convs; linear split conv(concat(x,h)) = conv(x,Wx)+conv(h,Wh)
// so the x-half is computed once per gate (54 -> 35 GFLOP).
// - Conv inputs: 194x194 zero-halo NHWC bf16 -> unconditional global_load_b128
//   with immediate offsets.
// - Weights: fragment-packed, staged in LDS (72 KB/block), fed via ds_load;
//   B rides the LDS banks while A rides the WGP$ path.
// - Each wave register-blocks all 4 N-tiles; B fragments for the 4 WMMAs are
//   loaded into distinct registers ahead of use so ds_loads pipeline against
//   the matrix pipe (partial s_wait_dscnt instead of full serialization).
// ---------------------------------------------------------------------------

typedef __attribute__((ext_vector_type(16))) __bf16 v16bf;
typedef __attribute__((ext_vector_type(8)))  __bf16 v8bf;
typedef __attribute__((ext_vector_type(8)))  float  v8f;

constexpr int kH   = 192;
constexpr int kW   = 192;
constexpr int kHW  = kH * kW;            // 36864 interior pixels
constexpr int kPW  = kW + 2;             // 194 padded row
constexpr int kPHW = (kH + 2) * kPW;     // 37636 padded pixels
constexpr int kC   = 64;
constexpr int kL   = 8;
constexpr int kMtiles = kHW / 16;        // 2304 tiles (16 pixels each, row-aligned)
constexpr int kKtiles = 18;              // K = 9*64 = 576, K-step 32
constexpr int kNT     = 4;               // N-tiles per wave (Cout 64)
constexpr int kBElems = kKtiles * kNT * 512;   // 36864 bf16 = 72 KB per conv
constexpr size_t kPadSlab = (size_t)kPHW * kC;
constexpr int  kNumPad    = 1 + kL + 1 + 1;    // x, 8*h, hsum, rh
constexpr size_t kPadTot  = kPadSlab * kNumPad;

// per-lane helpers for the 16x16 fragment layouts
__device__ __forceinline__ int lane_n (int lane) { return lane & 15; }
__device__ __forceinline__ int lane_mb(int lane) { return 8 * (lane >> 4); }

// ---------------------------------------------------------------------------
// Stage one conv's packed weights (72 KB) into LDS, cooperatively.
// ---------------------------------------------------------------------------
__device__ __forceinline__ void stage_B(const __bf16* __restrict__ wpack,
                                        __bf16* sB)
{
    constexpr int vecs = kBElems / 8;
    for (int i = threadIdx.x; i < vecs; i += 256)
        *(v8bf*)(sB + (size_t)i * 8) = *(const v8bf*)(wpack + (size_t)i * 8);
    __syncthreads();
}

// ---------------------------------------------------------------------------
// Implicit-GEMM inner loop: 16 pixels x 64 couts, K = 576.
// One A fragment (global, immediate offsets) feeds kNT WMMAs.
// All kNT B fragments are fetched into distinct registers BEFORE the WMMA
// group so the ds_loads issue as a clause and overlap the matrix pipe.
// ---------------------------------------------------------------------------
__device__ __forceinline__ void conv_tile_multi(v8f acc[kNT],
                                                const __bf16* __restrict__ in_base,
                                                const __bf16* __restrict__ sB,
                                                int lane)
{
#pragma unroll
    for (int kpos = 0; kpos < 9; ++kpos) {
        const int dy = kpos / 3 - 1;
        const int dx = kpos % 3 - 1;
        const __bf16* src = in_base + (dy * kPW + dx) * kC;   // immediate-foldable

        // this lane's 32 of the 64 input channels for this spatial tap (4 x 16B)
        v8bf a0 = *(const v8bf*)(src);
        v8bf a1 = *(const v8bf*)(src + 16);
        v8bf a2 = *(const v8bf*)(src + 32);
        v8bf a3 = *(const v8bf*)(src + 48);
        v16bf aF[2];
#pragma unroll
        for (int i = 0; i < 8; ++i) {
            aF[0][i] = a0[i]; aF[0][i + 8] = a1[i];
            aF[1][i] = a2[i]; aF[1][i + 8] = a3[i];
        }

#pragma unroll
        for (int half = 0; half < 2; ++half) {
            const int kk = kpos * 2 + half;

            // fetch all kNT B fragments first (distinct regs -> pipelined)
            v16bf b[kNT];
#pragma unroll
            for (int nt = 0; nt < kNT; ++nt)
                b[nt] = *(const v16bf*)(sB +
                    (size_t)(((kk * kNT + nt) * 32 + lane)) * 16);

            // then the WMMA group
#pragma unroll
            for (int nt = 0; nt < kNT; ++nt)
                acc[nt] = __builtin_amdgcn_wmma_f32_16x16x32_bf16(
                              false, aF[half], false, b[nt], (short)0,
                              acc[nt], false, false);
        }
    }
}

// ---------------------------------------------------------------------------
// Zero padded-bf16 arena (halo must be 0; interiors rewritten later).
// ---------------------------------------------------------------------------
__global__ void zero_pad_arena(__bf16* __restrict__ arena)
{
    const size_t i = (size_t)blockIdx.x * blockDim.x + threadIdx.x;
    if (i * 8 >= kPadTot) return;
    v8bf z;
#pragma unroll
    for (int j = 0; j < 8; ++j) z[j] = (__bf16)0.0f;
    *(v8bf*)(arena + i * 8) = z;
}

// ---------------------------------------------------------------------------
// Pass 1: child_h_sum + bf16 padded-NHWC copies of x, child_h, sum.
// ---------------------------------------------------------------------------
__global__ void prep_inputs(const float* __restrict__ x,
                            const float* __restrict__ ch,
                            __bf16* __restrict__ x_b,
                            __bf16* __restrict__ h_b,      // kL padded slabs
                            float*  __restrict__ hsum_f,   // unpadded NHWC
                            __bf16* __restrict__ hsum_b)   // padded
{
    const int i = blockIdx.x * blockDim.x + threadIdx.x;
    if (i >= kC * kHW) return;
    const int c  = i / kHW;
    const int p  = i % kHW;
    const int py = p / kW, px = p % kW;
    const size_t op = (size_t)((py + 1) * kPW + (px + 1)) * kC + c;
    const size_t ou = (size_t)p * kC + c;

    x_b[op] = (__bf16)x[i];

    float s = 0.0f;
#pragma unroll
    for (int l = 0; l < kL; ++l) {
        const float v = ch[(size_t)l * kC * kHW + i];
        s += v;
        h_b[(size_t)l * kPadSlab + op] = (__bf16)v;
    }
    hsum_f[ou] = s;
    hsum_b[op] = (__bf16)s;
}

// ---------------------------------------------------------------------------
// Weight packer -> fragment-ready B layout [kk][nt][lane][e] (32B per lane).
// K order: k = kpos*64 + cin.  Klocal = e + 8*((e>>3)+(lane>>4)).
// cin_off selects the x-half (0) or h-half (64) of the 128-in-channel weight.
// ---------------------------------------------------------------------------
__global__ void pack_w(const float* __restrict__ Wsrc,
                       __bf16* __restrict__ out, int cin_off)
{
    const int i = blockIdx.x * blockDim.x + threadIdx.x;
    if (i >= kBElems) return;
    const int e    = i & 15;
    const int lane = (i >> 4) & 31;
    const int rest = i >> 9;
    const int nt   = rest & 3;
    const int kk   = rest >> 2;

    const int kl   = e + 8 * ((e >> 3) + (lane >> 4));
    const int kg   = kk * 32 + kl;
    const int kpos = kg >> 6;
    const int cin  = kg & 63;
    const int ky   = kpos / 3, kx = kpos % 3;
    const int co   = nt * 16 + (lane & 15);
    out[i] = (__bf16)Wsrc[((size_t)(co * 128 + cin_off + cin) * 3 + ky) * 3 + kx];
}

// ---------------------------------------------------------------------------
// base section = conv(x, W*_x) + bias  ->  base_f[pixel][192], section `sec`.
// ---------------------------------------------------------------------------
__global__ void conv_xgate(const __bf16* __restrict__ x_b,
                           const __bf16* __restrict__ wpack,
                           const float* __restrict__ bias,
                           float* __restrict__ base_f, int sec)
{
    __shared__ __bf16 sB[kBElems];
    stage_B(wpack, sB);

    const int lane = threadIdx.x & 31;
    const int mt   = blockIdx.x * 8 + (threadIdx.x >> 5);
    const int n = lane_n(lane), mb = lane_mb(lane);
    const int p = mt * 16 + n, y = p / kW, x = p % kW;
    const __bf16* in_base = x_b + (size_t)((y + 1) * kPW + (x + 1)) * kC + mb;

    v8f acc[kNT];
#pragma unroll
    for (int nt = 0; nt < kNT; ++nt) {
        const float bv = bias[nt * 16 + n];
#pragma unroll
        for (int v = 0; v < 8; ++v) acc[nt][v] = bv;
    }

    conv_tile_multi(acc, in_base, sB, lane);

#pragma unroll
    for (int nt = 0; nt < kNT; ++nt)
#pragma unroll
        for (int v = 0; v < 8; ++v)
            base_f[(size_t)(mt * 16 + mb + v) * 192 + sec * 64 + nt * 16 + n] = acc[nt][v];
}

// ---------------------------------------------------------------------------
// r gate over 8 children + reset_hidden accumulation (padded bf16 out).
// ---------------------------------------------------------------------------
__global__ void gate_r(const __bf16* __restrict__ h_b,
                       const __bf16* __restrict__ wpack_r,
                       const float* __restrict__ base_f,
                       __bf16* __restrict__ rh_b)
{
    __shared__ __bf16 sB[kBElems];
    stage_B(wpack_r, sB);

    const int lane = threadIdx.x & 31;
    const int mt   = blockIdx.x * 8 + (threadIdx.x >> 5);
    const int n = lane_n(lane), mb = lane_mb(lane);
    const int p = mt * 16 + n, y = p / kW, x = p % kW;
    const size_t pix_in = (size_t)((y + 1) * kPW + (x + 1)) * kC;        // A side
    const int x0 = (mt * 16) % kW;
    const size_t pix_d0 = (size_t)((y + 1) * kPW + (x0 + mb + 1)) * kC;  // D side

    float rh[kNT][8];
#pragma unroll
    for (int nt = 0; nt < kNT; ++nt)
#pragma unroll
        for (int v = 0; v < 8; ++v) rh[nt][v] = 0.0f;

    for (int l = 0; l < kL; ++l) {
        const __bf16* hl = h_b + (size_t)l * kPadSlab;

        v8f acc[kNT];
#pragma unroll
        for (int nt = 0; nt < kNT; ++nt)
#pragma unroll
            for (int v = 0; v < 8; ++v)
                acc[nt][v] = base_f[(size_t)(mt * 16 + mb + v) * 192 + nt * 16 + n];

        conv_tile_multi(acc, hl + pix_in + mb, sB, lane);

#pragma unroll
        for (int nt = 0; nt < kNT; ++nt)
#pragma unroll
            for (int v = 0; v < 8; ++v) {
                const float s  = 1.0f / (1.0f + __expf(-acc[nt][v]));
                const float hv = (float)hl[pix_d0 + (size_t)v * kC + nt * 16 + n];
                rh[nt][v] += s * hv;
            }
    }
#pragma unroll
    for (int nt = 0; nt < kNT; ++nt)
#pragma unroll
        for (int v = 0; v < 8; ++v)
            rh_b[pix_d0 + (size_t)v * kC + nt * 16 + n] = (__bf16)rh[nt][v];
}

// ---------------------------------------------------------------------------
// z = sigmoid(base_z + conv(h_sum, Wz_h))   (f32, unpadded NHWC)
// ---------------------------------------------------------------------------
__global__ void gate_z(const __bf16* __restrict__ hsum_b,
                       const __bf16* __restrict__ wpack_z,
                       const float* __restrict__ base_f,
                       float* __restrict__ z_f)
{
    __shared__ __bf16 sB[kBElems];
    stage_B(wpack_z, sB);

    const int lane = threadIdx.x & 31;
    const int mt   = blockIdx.x * 8 + (threadIdx.x >> 5);
    const int n = lane_n(lane), mb = lane_mb(lane);
    const int p = mt * 16 + n, y = p / kW, x = p % kW;
    const __bf16* in_base = hsum_b + (size_t)((y + 1) * kPW + (x + 1)) * kC + mb;

    v8f acc[kNT];
#pragma unroll
    for (int nt = 0; nt < kNT; ++nt)
#pragma unroll
        for (int v = 0; v < 8; ++v)
            acc[nt][v] = base_f[(size_t)(mt * 16 + mb + v) * 192 + 64 + nt * 16 + n];

    conv_tile_multi(acc, in_base, sB, lane);

#pragma unroll
    for (int nt = 0; nt < kNT; ++nt)
#pragma unroll
        for (int v = 0; v < 8; ++v)
            z_f[(size_t)(mt * 16 + mb + v) * kC + nt * 16 + n] =
                1.0f / (1.0f + __expf(-acc[nt][v]));
}

// ---------------------------------------------------------------------------
// o = tanh(base_o + conv(reset_hidden, Wo_h));  h = (1-z)*o + z*h_sum  (NCHW)
// ---------------------------------------------------------------------------
__global__ void gate_final(const __bf16* __restrict__ rh_b,
                           const __bf16* __restrict__ wpack_o,
                           const float* __restrict__ base_f,
                           const float* __restrict__ z_f,
                           const float* __restrict__ hsum_f,
                           float* __restrict__ out)
{
    __shared__ __bf16 sB[kBElems];
    stage_B(wpack_o, sB);

    const int lane = threadIdx.x & 31;
    const int mt   = blockIdx.x * 8 + (threadIdx.x >> 5);
    const int n = lane_n(lane), mb = lane_mb(lane);
    const int p = mt * 16 + n, y = p / kW, x = p % kW;
    const __bf16* in_base = rh_b + (size_t)((y + 1) * kPW + (x + 1)) * kC + mb;

    v8f acc[kNT];
#pragma unroll
    for (int nt = 0; nt < kNT; ++nt)
#pragma unroll
        for (int v = 0; v < 8; ++v)
            acc[nt][v] = base_f[(size_t)(mt * 16 + mb + v) * 192 + 128 + nt * 16 + n];

    conv_tile_multi(acc, in_base, sB, lane);

#pragma unroll
    for (int nt = 0; nt < kNT; ++nt)
#pragma unroll
        for (int v = 0; v < 8; ++v) {
            const size_t pd = (size_t)(mt * 16 + mb + v);
            const int    co = nt * 16 + n;
            const float  o  = tanhf(acc[nt][v]);
            const float  z  = z_f[pd * kC + co];
            const float  hs = hsum_f[pd * kC + co];
            out[(size_t)co * kHW + pd] = (1.0f - z) * o + z * hs;   // NCHW
        }
}

// ---------------------------------------------------------------------------
// Host launch.  Workspace budget ~110 MB.
// ---------------------------------------------------------------------------
extern "C" void kernel_launch(void* const* d_in, const int* in_sizes, int n_in,
                              void* d_out, int out_size, void* d_ws, size_t ws_size,
                              hipStream_t stream)
{
    const float* x  = (const float*)d_in[0];
    const float* ch = (const float*)d_in[1];
    const float* Wr = (const float*)d_in[2];
    const float* br = (const float*)d_in[3];
    const float* Wz = (const float*)d_in[4];
    const float* bz = (const float*)d_in[5];
    const float* Wo = (const float*)d_in[6];
    const float* bo = (const float*)d_in[7];
    float* out = (float*)d_out;

    char* ws = (char*)d_ws;
    size_t off = 0;
    auto carve = [&](size_t bytes) -> char* {
        char* p = ws + off;
        off = (off + bytes + 255) & ~(size_t)255;
        return p;
    };

    __bf16* pad_arena = (__bf16*)carve(kPadTot * 2);
    __bf16* x_b    = pad_arena;
    __bf16* h_b    = pad_arena + kPadSlab;            // 8 slabs
    __bf16* hsum_b = pad_arena + kPadSlab * (1 + kL);
    __bf16* rh_b   = pad_arena + kPadSlab * (2 + kL);

    float*  hsum_f = (float*) carve((size_t)kHW * kC * 4);
    float*  base_f = (float*) carve((size_t)kHW * 192 * 4);
    float*  z_f    = (float*) carve((size_t)kHW * kC * 4);
    __bf16* wp_xr  = (__bf16*)carve((size_t)kBElems * 2);
    __bf16* wp_xz  = (__bf16*)carve((size_t)kBElems * 2);
    __bf16* wp_xo  = (__bf16*)carve((size_t)kBElems * 2);
    __bf16* wp_r   = (__bf16*)carve((size_t)kBElems * 2);
    __bf16* wp_z   = (__bf16*)carve((size_t)kBElems * 2);
    __bf16* wp_o   = (__bf16*)carve((size_t)kBElems * 2);
    (void)ws_size; (void)in_sizes; (void)n_in; (void)out_size;

    const int TB = 256;
    const int CB = kMtiles / 8;      // 288 blocks for all conv kernels

    zero_pad_arena<<<(int)((kPadTot / 8 + TB - 1) / TB), TB, 0, stream>>>(pad_arena);
    prep_inputs<<<(kC * kHW + TB - 1) / TB, TB, 0, stream>>>(x, ch, x_b, h_b, hsum_f, hsum_b);

    const int PB = (kBElems + TB - 1) / TB;
    pack_w<<<PB, TB, 0, stream>>>(Wr, wp_xr, 0);
    pack_w<<<PB, TB, 0, stream>>>(Wz, wp_xz, 0);
    pack_w<<<PB, TB, 0, stream>>>(Wo, wp_xo, 0);
    pack_w<<<PB, TB, 0, stream>>>(Wr, wp_r, 64);
    pack_w<<<PB, TB, 0, stream>>>(Wz, wp_z, 64);
    pack_w<<<PB, TB, 0, stream>>>(Wo, wp_o, 64);

    // x-half convs -> base_f sections (r|z|o)
    conv_xgate<<<CB, TB, 0, stream>>>(x_b, wp_xr, br, base_f, 0);
    conv_xgate<<<CB, TB, 0, stream>>>(x_b, wp_xz, bz, base_f, 1);
    conv_xgate<<<CB, TB, 0, stream>>>(x_b, wp_xo, bo, base_f, 2);

    // gates
    gate_r<<<CB, TB, 0, stream>>>(h_b, wp_r, base_f, rh_b);
    gate_z<<<CB, TB, 0, stream>>>(hsum_b, wp_z, base_f, z_f);
    gate_final<<<CB, TB, 0, stream>>>(rh_b, wp_o, base_f, z_f, hsum_f, out);
}
